// MultiHead_Self_Attention_51943334478448
// MI455X (gfx1250) — matrix operand order repaired and checked by reading, not compiled, not run on hardware
//
#include <hip/hip_runtime.h>
#include <hip/hip_bf16.h>

typedef __attribute__((ext_vector_type(16))) _Float16 v16h;
typedef __attribute__((ext_vector_type(8)))  _Float16 v8h;
typedef __attribute__((ext_vector_type(4)))  _Float16 v4h;
typedef __attribute__((ext_vector_type(8)))  float    v8f;

#define WMMA_F16(a, b, c) \
  __builtin_amdgcn_wmma_f32_16x16x32_f16(false, (a), false, (b), (short)0, (c), false, false)

#define CAT16(lo, hi) \
  __builtin_shufflevector((lo), (hi), 0, 1, 2, 3, 4, 5, 6, 7, 8, 9, 10, 11, 12, 13, 14, 15)

// Problem constants (B=2, S=2048, D=1024, H=16, dk=64)
#define SB   2
#define SS   2048
#define SD   1024
#define SH   16
#define SDK  64

// ---------------------------------------------------------------------------
// Kernel 0: bulk f32 -> f16 conversion (memory-bound one-shot pass).
// ---------------------------------------------------------------------------
__global__ __launch_bounds__(256) void f32_to_f16_kernel(
    const float* __restrict__ src, _Float16* __restrict__ dst, int n4)
{
  const int i = blockIdx.x * blockDim.x + threadIdx.x;
  if (i < n4) {
    const float4 v = ((const float4*)src)[i];
    v4h h;
    h[0] = (_Float16)v.x; h[1] = (_Float16)v.y;
    h[2] = (_Float16)v.z; h[3] = (_Float16)v.w;
    ((v4h*)dst)[i] = h;
  }
}

// ---------------------------------------------------------------------------
// Kernel 1: fused QKV projection (x @ W^T) + RoPE, all-f16 operands.
//   Q,K -> [B,H,S,64] (row-major in dk), V -> [B,H,64,S] (transposed)
// Each wave: 32 rows x 16 cols; 2 A fragments reuse 3 B fragments per step
// (10 b128 loads per 6 WMMAs).
// ---------------------------------------------------------------------------
__global__ __launch_bounds__(128) void qkv_rope_kernel(
    const _Float16* __restrict__ xh, const int* __restrict__ tp,
    const _Float16* __restrict__ wqh, const _Float16* __restrict__ wkh,
    const _Float16* __restrict__ wvh,
    _Float16* __restrict__ Q, _Float16* __restrict__ K,
    _Float16* __restrict__ Vt)
{
  const int lane = threadIdx.x & 31;
  const int wave = threadIdx.x >> 5;
  const int l  = lane & 15;
  const int hi = lane >> 4;
  const int m0 = blockIdx.x * 32;                    // 32-row tile in [0, B*S)
  const int n0 = (blockIdx.y * 4 + wave) * 16;       // col tile in [0, D)

  const _Float16* xrow0 = xh  + (size_t)(m0 + l) * SD;
  const _Float16* xrow1 = xh  + (size_t)(m0 + 16 + l) * SD;
  const _Float16* wqrow = wqh + (size_t)(n0 + l) * SD;
  const _Float16* wkrow = wkh + (size_t)(n0 + l) * SD;
  const _Float16* wvrow = wvh + (size_t)(n0 + l) * SD;

  v8f accq[2] = {}, acck[2] = {}, accv[2] = {};

  for (int k0 = 0; k0 < SD; k0 += 32) {
    __builtin_prefetch(xrow0 + k0 + 128, 0, 0);
    // A fragments: lanes 0-15 K=0..15, lanes 16-31 K=16..31
    const v16h a0 = *(const v16h*)(xrow0 + k0 + hi * 16);
    const v16h a1 = *(const v16h*)(xrow1 + k0 + hi * 16);
    // B fragments: lane n = col n; lanes 0-15 K=0..15, lanes 16-31 K=16..31
    const v16h bq = *(const v16h*)(wqrow + k0 + hi * 16);
    const v16h bk = *(const v16h*)(wkrow + k0 + hi * 16);
    const v16h bv = *(const v16h*)(wvrow + k0 + hi * 16);
    accq[0] = WMMA_F16(a0, bq, accq[0]);
    accq[1] = WMMA_F16(a1, bq, accq[1]);
    acck[0] = WMMA_F16(a0, bk, acck[0]);
    acck[1] = WMMA_F16(a1, bk, acck[1]);
    accv[0] = WMMA_F16(a0, bv, accv[0]);
    accv[1] = WMMA_F16(a1, bv, accv[1]);
  }

  // RoPE on Q,K + scatter stores. C layout: lane -> col, reg r -> row (hi*8+r).
  const int e   = n0 + l;
  const int h   = e >> 6;
  const int rh  = e & 63;
  const int j   = rh >> 1;
  const bool odd = (rh & 1);
  // inv_freq = 10000^(-j/32)
  const float inv_freq = __expf(-(float)j * 0.2878231366242557f);

#pragma unroll
  for (int g = 0; g < 2; ++g) {
#pragma unroll
    for (int r = 0; r < 8; ++r) {
      const int row = m0 + g * 16 + hi * 8 + r;
      const int s   = row & (SS - 1);
      const int b   = row >> 11;
      const float pos = (float)tp[s];
      float sn, cs;
      __sincosf(pos * inv_freq, &sn, &cs);

      const float sq = accq[g][r];
      const float pq = __shfl_xor(sq, 1, 32);   // partner column e^1
      const float oq = odd ? (sn * pq + cs * sq) : (cs * sq - sn * pq);

      const float sk = acck[g][r];
      const float pk = __shfl_xor(sk, 1, 32);
      const float ok = odd ? (sn * pk + cs * sk) : (cs * sk - sn * pk);

      const size_t qidx = ((((size_t)b * SH + h) * SS) + s) * SDK + rh;
      Q[qidx] = (_Float16)oq;
      K[qidx] = (_Float16)ok;
      const size_t vidx = (((size_t)b * SH + h) * SDK + rh) * SS + s;  // transposed
      Vt[vidx] = (_Float16)accv[g][r];
    }
  }
}

// ---------------------------------------------------------------------------
// Kernel 2: causal flash attention. One wave owns 16 query rows of one (b,h).
// 32 keys/step: 4 score WMMAs -> online softmax -> P staged through LDS
// (C-layout -> A-layout) -> 4 PV WMMAs into a 16x64 f32 accumulator.
// ---------------------------------------------------------------------------
__global__ __launch_bounds__(256) void attn_kernel(
    const _Float16* __restrict__ Q, const _Float16* __restrict__ K,
    const _Float16* __restrict__ Vt, _Float16* __restrict__ O)
{
  __shared__ alignas(32) _Float16 plds[8][16 * 32];   // 1KB per wave
  const int lane = threadIdx.x & 31;
  const int wave = threadIdx.x >> 5;
  const int l  = lane & 15;
  const int hi = lane >> 4;
  const int b  = blockIdx.z;
  const int h  = blockIdx.y;
  const int m0 = (blockIdx.x * 8 + wave) * 16;

  const _Float16* Qh = Q  + (((size_t)b * SH + h) * SS) * SDK;
  const _Float16* Kh = K  + (((size_t)b * SH + h) * SS) * SDK;
  const _Float16* Vh = Vt + (((size_t)b * SH + h) * SDK) * SS;

  // Q A-fragments for dk 0..31 and 32..63 (resident in registers)
  v16h qa0, qa1;
  {
    const _Float16* qrow = Qh + (size_t)(m0 + l) * SDK;
    qa0 = *(const v16h*)(qrow + hi * 16);
    qa1 = *(const v16h*)(qrow + 32 + hi * 16);
  }

  v8f o0 = {}, o1 = {}, o2 = {}, o3 = {};
  float m_i[8], l_i[8];
#pragma unroll
  for (int r = 0; r < 8; ++r) { m_i[r] = -1e30f; l_i[r] = 0.f; }

  const int nsteps = (m0 + 16 + 31) >> 5;
  for (int kt = 0; kt < nsteps; ++kt) {
    const int kb = kt << 5;

    // ---- scores: two 16-key tiles, each = 2 WMMAs over dk ----
    v8f s0 = {}, s1 = {};
    {
      const _Float16* kr0 = Kh + (size_t)(kb + l) * SDK;
      const _Float16* kr1 = Kh + (size_t)(kb + 16 + l) * SDK;
      const v16h b0l = *(const v16h*)(kr0 + hi * 16);
      const v16h b0h = *(const v16h*)(kr0 + 32 + hi * 16);
      const v16h b1l = *(const v16h*)(kr1 + hi * 16);
      const v16h b1h = *(const v16h*)(kr1 + 32 + hi * 16);
      s0 = WMMA_F16(qa0, b0l, s0);
      s0 = WMMA_F16(qa1, b0h, s0);
      s1 = WMMA_F16(qa0, b1l, s1);
      s1 = WMMA_F16(qa1, b1h, s1);
    }

    // ---- online softmax update (rows live in 16-lane halves) ----
    __threadfence_block();   // order prior-iteration LDS reads before rewrite
#pragma unroll
    for (int r = 0; r < 8; ++r) {
      const int qi = m0 + hi * 8 + r;
      float v0 = s0[r] * 0.125f;               // 1/sqrt(64)
      float v1 = s1[r] * 0.125f;
      if (kb + l > qi)      v0 = -3e38f;       // causal mask
      if (kb + 16 + l > qi) v1 = -3e38f;

      float mx = fmaxf(v0, v1);
      mx = fmaxf(mx, __shfl_xor(mx, 1, 32));
      mx = fmaxf(mx, __shfl_xor(mx, 2, 32));
      mx = fmaxf(mx, __shfl_xor(mx, 4, 32));
      mx = fmaxf(mx, __shfl_xor(mx, 8, 32));
      const float mnew  = fmaxf(m_i[r], mx);
      const float alpha = __expf(m_i[r] - mnew);
      const float p0 = __expf(v0 - mnew);
      const float p1 = __expf(v1 - mnew);

      float rs = p0 + p1;
      rs += __shfl_xor(rs, 1, 32);
      rs += __shfl_xor(rs, 2, 32);
      rs += __shfl_xor(rs, 4, 32);
      rs += __shfl_xor(rs, 8, 32);

      l_i[r] = l_i[r] * alpha + rs;
      m_i[r] = mnew;
      o0[r] *= alpha; o1[r] *= alpha; o2[r] *= alpha; o3[r] *= alpha;

      plds[wave][(hi * 8 + r) * 32 + l]      = (_Float16)p0;
      plds[wave][(hi * 8 + r) * 32 + 16 + l] = (_Float16)p1;
    }
    __threadfence_block();   // LDS stores visible before A-layout reload

    // ---- reload P as 16x32 A-fragment (ds_load_b128 x2) ----
    const _Float16* prow = &plds[wave][l * 32];
    const v16h pa = *(const v16h*)(prow + hi * 16);

    // ---- P @ V : 4 output column tiles of dk ----
    const _Float16* vbase = Vh + (size_t)l * SS + kb + hi * 16;
    const v16h vb0 = *(const v16h*)(vbase + (size_t)0  * SS);
    const v16h vb1 = *(const v16h*)(vbase + (size_t)16 * SS);
    const v16h vb2 = *(const v16h*)(vbase + (size_t)32 * SS);
    const v16h vb3 = *(const v16h*)(vbase + (size_t)48 * SS);
    o0 = WMMA_F16(pa, vb0, o0);
    o1 = WMMA_F16(pa, vb1, o1);
    o2 = WMMA_F16(pa, vb2, o2);
    o3 = WMMA_F16(pa, vb3, o3);
  }

  // ---- normalize and store O[b, s, h*64 + e] as f16 ----
#pragma unroll
  for (int r = 0; r < 8; ++r) {
    const float inv = 1.0f / l_i[r];
    const int s = m0 + hi * 8 + r;
    const size_t base = ((size_t)b * SS + s) * SD + h * SDK;
    O[base +  0 + l] = (_Float16)(o0[r] * inv);
    O[base + 16 + l] = (_Float16)(o1[r] * inv);
    O[base + 32 + l] = (_Float16)(o2[r] * inv);
    O[base + 48 + l] = (_Float16)(o3[r] * inv);
  }
}

// ---------------------------------------------------------------------------
// Kernel 3: output projection  out[b,s,d] = sum_e O[b,s,e] * wo[d,e]  (f32 out)
// Each wave: 32 rows x 32 cols (4 WMMAs per step, 8 b128 loads).
// ---------------------------------------------------------------------------
__global__ __launch_bounds__(128) void out_proj_kernel(
    const _Float16* __restrict__ O, const _Float16* __restrict__ woh,
    float* __restrict__ out)
{
  const int lane = threadIdx.x & 31;
  const int wave = threadIdx.x >> 5;
  const int l  = lane & 15;
  const int hi = lane >> 4;
  const int m0 = blockIdx.x * 32;
  const int n0 = (blockIdx.y * 4 + wave) * 32;

  const _Float16* orow0 = O   + (size_t)(m0 + l) * SD;
  const _Float16* orow1 = O   + (size_t)(m0 + 16 + l) * SD;
  const _Float16* wrow0 = woh + (size_t)(n0 + l) * SD;
  const _Float16* wrow1 = woh + (size_t)(n0 + 16 + l) * SD;

  v8f acc00 = {}, acc01 = {}, acc10 = {}, acc11 = {};
  for (int k0 = 0; k0 < SD; k0 += 32) {
    const v16h a0 = *(const v16h*)(orow0 + k0 + hi * 16);
    const v16h a1 = *(const v16h*)(orow1 + k0 + hi * 16);
    const v16h b0 = *(const v16h*)(wrow0 + k0 + hi * 16);
    const v16h b1 = *(const v16h*)(wrow1 + k0 + hi * 16);
    acc00 = WMMA_F16(a0, b0, acc00);
    acc01 = WMMA_F16(a0, b1, acc01);
    acc10 = WMMA_F16(a1, b0, acc10);
    acc11 = WMMA_F16(a1, b1, acc11);
  }

#pragma unroll
  for (int r = 0; r < 8; ++r) {
    const size_t row0 = (size_t)(m0 + hi * 8 + r) * SD;
    const size_t row1 = (size_t)(m0 + 16 + hi * 8 + r) * SD;
    out[row0 + n0 + l]      = acc00[r];
    out[row0 + n0 + 16 + l] = acc01[r];
    out[row1 + n0 + l]      = acc10[r];
    out[row1 + n0 + 16 + l] = acc11[r];
  }
}

// ---------------------------------------------------------------------------
extern "C" void kernel_launch(void* const* d_in, const int* in_sizes, int n_in,
                              void* d_out, int out_size, void* d_ws, size_t ws_size,
                              hipStream_t stream) {
  const float* x  = (const float*)d_in[0];
  const int*   tp = (const int*)d_in[1];
  const float* wq = (const float*)d_in[2];
  const float* wk = (const float*)d_in[3];
  const float* wv = (const float*)d_in[4];
  const float* wo = (const float*)d_in[5];
  float* out = (float*)d_out;

  const size_t NBIG = (size_t)SB * SH * SS * SDK;   // 4,194,304 elems
  const size_t NW   = (size_t)SD * SD;              // 1,048,576 elems
  _Float16* Qh  = (_Float16*)d_ws;
  _Float16* Kh  = Qh  + NBIG;
  _Float16* Vt  = Kh  + NBIG;
  _Float16* Oh  = Vt  + NBIG;       // [B,S,D] f16
  _Float16* Xh  = Oh  + NBIG;       // [B*S,D] f16
  _Float16* Wqh = Xh  + NBIG;
  _Float16* Wkh = Wqh + NW;
  _Float16* Wvh = Wkh + NW;
  _Float16* Woh = Wvh + NW;         // total 48 MB of d_ws

  // One-shot f32 -> f16 conversions (memory-bound)
  f32_to_f16_kernel<<<(int)(NBIG / 4 / 256), 256, 0, stream>>>(x,  Xh,  (int)(NBIG / 4));
  f32_to_f16_kernel<<<(int)(NW   / 4 / 256), 256, 0, stream>>>(wq, Wqh, (int)(NW / 4));
  f32_to_f16_kernel<<<(int)(NW   / 4 / 256), 256, 0, stream>>>(wk, Wkh, (int)(NW / 4));
  f32_to_f16_kernel<<<(int)(NW   / 4 / 256), 256, 0, stream>>>(wv, Wvh, (int)(NW / 4));
  f32_to_f16_kernel<<<(int)(NW   / 4 / 256), 256, 0, stream>>>(wo, Woh, (int)(NW / 4));

  // QKV + RoPE: 128 (32-row) tiles x (16 blocks * 4 waves) col tiles
  qkv_rope_kernel<<<dim3((SB * SS) / 32, SD / 64), 128, 0, stream>>>(
      Xh, tp, Wqh, Wkh, Wvh, Qh, Kh, Vt);

  // Flash attention: 16 q-tiles/block * 8 waves, per (h, b)
  attn_kernel<<<dim3(SS / 128, SH, SB), 256, 0, stream>>>(Qh, Kh, Vt, Oh);

  // Output projection: 128 (32-row) tiles x (8 blocks * 4 waves * 32 cols)
  out_proj_kernel<<<dim3((SB * SS) / 32, SD / 128), 128, 0, stream>>>(Oh, Woh, out);
}
